// STUCore_49417893708193
// MI455X (gfx1250) — matrix operand order repaired and checked by reading, not compiled, not run on hardware
//
#include <hip/hip_runtime.h>

typedef __attribute__((ext_vector_type(16))) __bf16       v16bf;
typedef __attribute__((ext_vector_type(8)))  float        v8f;
typedef __attribute__((ext_vector_type(4)))  unsigned int u32x4;
typedef __attribute__((ext_vector_type(4)))  int          i32x4;

#define D_MODEL 1024
#define SEQ_LEN 2048
#define BATCH   4
#define NFFT    4096
#define MROWS   (BATCH * SEQ_LEN)   // 8192

struct c32 { float x, y; };

// ---- gfx1250 async global->LDS path (guarded; falls back to sync copy) ----
#if defined(__has_builtin)
#  if __has_builtin(__builtin_amdgcn_global_load_async_to_lds_b128)
#    define HAVE_ASYNC_LDS 1
#  else
#    define HAVE_ASYNC_LDS 0
#  endif
#  if __has_builtin(__builtin_amdgcn_s_wait_asynccnt)
#    define HAVE_WAIT_ASYNC 1
#  else
#    define HAVE_WAIT_ASYNC 0
#  endif
#else
#  define HAVE_ASYNC_LDS 0
#  define HAVE_WAIT_ASYNC 0
#endif

static __device__ __forceinline__ void async_copy16(const unsigned short* g,
                                                    unsigned short* l) {
#if HAVE_ASYNC_LDS
  __builtin_amdgcn_global_load_async_to_lds_b128(
      (__attribute__((address_space(1))) i32x4*)g,
      (__attribute__((address_space(3))) i32x4*)l, 0, 0);
#else
  *reinterpret_cast<u32x4*>(l) = *reinterpret_cast<const u32x4*>(g);
#endif
}

static __device__ __forceinline__ void wait_async0() {
#if HAVE_WAIT_ASYNC
  __builtin_amdgcn_s_wait_asynccnt(0);
#else
  asm volatile("s_wait_asynccnt 0" ::: "memory");
#endif
}

static __device__ __forceinline__ unsigned short f32_to_bf16(float f) {
  unsigned int u = __float_as_uint(f);
  unsigned int r = u + 0x7FFFu + ((u >> 16) & 1u);   // round-to-nearest-even
  return (unsigned short)(r >> 16);
}

// ---------------------------------------------------------------------------
// fp32 -> bf16 conversion
// ---------------------------------------------------------------------------
__global__ void cvt_bf16_kernel(const float* __restrict__ in,
                                unsigned short* __restrict__ out, int n) {
  int i = blockIdx.x * blockDim.x + threadIdx.x;
  if (i < n) out[i] = f32_to_bf16(in[i]);
}

// out[n*1024 + k] = bf16(in[k*1024 + n])  (pre-transpose M_inputs -> [N][K])
__global__ void cvt_transpose_bf16_kernel(const float* __restrict__ in,
                                          unsigned short* __restrict__ out) {
  int idx = blockIdx.x * blockDim.x + threadIdx.x;
  int k = idx & (D_MODEL - 1);
  int n = idx >> 10;
  out[idx] = f32_to_bf16(in[(size_t)k * D_MODEL + n]);
}

// ---------------------------------------------------------------------------
// x_proj = x @ M_inputs   (bf16 WMMA, fp32 accumulate)
// A:  [8192,1024] bf16 row-major ([m][k])
// Bt: [1024,1024] bf16, pre-transposed ([n][k]) -> contiguous async staging
// Output channel-major: Ct[(b*D + n)*L + t], m = b*L + t
// Block tile 128x128, 4 waves (wave32), each wave 64x64 = 4x4 WMMA tiles.
// Double-buffered LDS with GLOBAL_LOAD_ASYNC_TO_LDS_B128 (ASYNCcnt pipeline).
// ---------------------------------------------------------------------------
__global__ __launch_bounds__(128)
void gemm_xproj_wmma(const unsigned short* __restrict__ A,
                     const unsigned short* __restrict__ Bt,
                     float* __restrict__ Ct) {
  __shared__ alignas(16) unsigned short As[2][128][32];   // [buf][m][k]
  __shared__ alignas(16) unsigned short Bts[2][128][32];  // [buf][n][k]

  const int tid  = threadIdx.x;
  const int lane = tid & 31;
  const int wave = tid >> 5;
  const int wr   = wave >> 1;         // 2x2 wave grid
  const int wc   = wave & 1;
  const int rowBase = blockIdx.x * 128;
  const int colBase = blockIdx.y * 128;

  const int half = lane >> 4;         // which 16-lane half of the wave
  const int l16  = lane & 15;

  v8f acc[4][4] = {};

  // per-thread staging coordinates: 512 16B-chunks per slab, 4 per thread
  int sr[4], sc8[4];
#pragma unroll
  for (int c = 0; c < 4; ++c) {
    int chunk = tid + c * 128;
    sr[c]  = chunk >> 2;
    sc8[c] = (chunk & 3) * 8;
  }

#define STAGE_SLABS(bufi, kk)                                                  \
  do {                                                                         \
    _Pragma("unroll")                                                          \
    for (int c = 0; c < 4; ++c) {                                              \
      async_copy16(&A[(size_t)(rowBase + sr[c]) * D_MODEL + (kk) + sc8[c]],    \
                   &As[bufi][sr[c]][sc8[c]]);                                  \
      async_copy16(&Bt[(size_t)(colBase + sr[c]) * D_MODEL + (kk) + sc8[c]],   \
                   &Bts[bufi][sr[c]][sc8[c]]);                                 \
    }                                                                          \
  } while (0)

  // prologue: stage K-slab 0
  STAGE_SLABS(0, 0);
  wait_async0();
  __syncthreads();

  for (int k0 = 0; k0 < D_MODEL; k0 += 32) {
    const int cur = (k0 >> 5) & 1;
    // issue async copies for next slab into the other buffer (safe: the
    // previous iteration's trailing barrier guarantees no reader remains)
    if (k0 + 32 < D_MODEL) STAGE_SLABS(cur ^ 1, k0 + 32);

    // --- build fragments per ISA 7.12.2 layouts ---
    v16bf aF[4], bF[4];
#pragma unroll
    for (int i = 0; i < 4; ++i) {
      int r = wr * 64 + i * 16 + l16;         // A row
      union { u32x4 q[2]; v16bf v; } u;
      u.q[0] = *reinterpret_cast<const u32x4*>(&As[cur][r][half * 8]);       // K = 8h..
      u.q[1] = *reinterpret_cast<const u32x4*>(&As[cur][r][16 + half * 8]);  // K = 16+8h..
      aF[i] = u.v;
    }
#pragma unroll
    for (int j = 0; j < 4; ++j) {
      int n = wc * 64 + j * 16 + l16;         // B column
      union { u32x4 q[2]; v16bf v; } u;
      u.q[0] = *reinterpret_cast<const u32x4*>(&Bts[cur][n][half * 8]);      // K = 8h..
      u.q[1] = *reinterpret_cast<const u32x4*>(&Bts[cur][n][16 + half * 8]); // K = 16+8h..
      bF[j] = u.v;
    }
#pragma unroll
    for (int i = 0; i < 4; ++i)
#pragma unroll
      for (int j = 0; j < 4; ++j)
        acc[i][j] = __builtin_amdgcn_wmma_f32_16x16x32_bf16(
            false, aF[i], false, bF[j], (short)0, acc[i][j], false, false);

    wait_async0();      // next slab resident in LDS (this wave's copies)
    __syncthreads();    // ... and everyone else's
  }
#undef STAGE_SLABS

  // --- store, channel-major [b, n, t] so the FFT stage reads contiguously ---
#pragma unroll
  for (int i = 0; i < 4; ++i)
#pragma unroll
    for (int j = 0; j < 4; ++j)
#pragma unroll
      for (int v = 0; v < 8; ++v) {
        int m = rowBase + wr * 64 + i * 16 + half * 8 + v;   // C/D: row = 8*half + vgpr
        int n = colBase + wc * 64 + j * 16 + l16;
        int b = m >> 11;
        int t = m & (SEQ_LEN - 1);
        Ct[(((size_t)b << 10) | n) * SEQ_LEN + t] = acc[i][j][v];
      }
}

// ---------------------------------------------------------------------------
// phi_proj: v[t,d] = sum_k phi[t,k] * Mf[k,d]   (tiny, VALU)
// ---------------------------------------------------------------------------
__global__ void phi_proj_kernel(const float* __restrict__ phi,
                                const float* __restrict__ Mf,
                                float* __restrict__ V) {
  int idx = blockIdx.x * blockDim.x + threadIdx.x;   // t*1024 + d
  int t = idx >> 10, d = idx & 1023;
  float s = 0.f;
#pragma unroll
  for (int k = 0; k < 24; ++k) s += phi[t * 24 + k] * Mf[k * D_MODEL + d];
  V[idx] = s;
}

// ---------------------------------------------------------------------------
// 4096-pt radix-2 Stockham FFT in LDS (natural in / natural out, ping-pong)
// 12 stages (even) -> result lands back in bufA. sign=-1 fwd, +1 inv.
// ---------------------------------------------------------------------------
static __device__ __forceinline__ void fft4096_stockham(c32* bufA, c32* bufB,
                                                        int tid, float sign) {
  c32* src = bufA;
  c32* dst = bufB;
#pragma unroll 1
  for (int s = 0; s < 12; ++s) {
    __syncthreads();
    const int Ls = 1 << s;
#pragma unroll 1
    for (int j = tid; j < NFFT / 2; j += 256) {
      int k = j & (Ls - 1);
      int g = j >> s;
      float ang = sign * 3.14159265358979f * (float)k / (float)Ls;
      float wr = __cosf(ang), wi = __sinf(ang);
      c32 a = src[j];
      c32 b = src[j + NFFT / 2];
      c32 t = { wr * b.x - wi * b.y, wr * b.y + wi * b.x };
      int o = (g << (s + 1)) + k;
      dst[o]      = c32{ a.x + t.x, a.y + t.y };
      dst[o + Ls] = c32{ a.x - t.x, a.y - t.y };
    }
    c32* tmp = src; src = dst; dst = tmp;
  }
  __syncthreads();
}

// ---------------------------------------------------------------------------
// V_f = FFT_4096 of the even-lag filter (odd taps zeroed), per channel d.
// plus+minus collapses to 2 * conv(u, v_even), so one spectrum suffices.
// ---------------------------------------------------------------------------
__global__ __launch_bounds__(256)
void vf_fft_kernel(const float* __restrict__ V, c32* __restrict__ Vf) {
  __shared__ c32 sA[NFFT];
  __shared__ c32 sB[NFFT];
  const int d = blockIdx.x, tid = threadIdx.x;
  for (int n = tid; n < NFFT; n += 256) {
    float val = (n < SEQ_LEN && ((n & 1) == 0)) ? V[(size_t)n * D_MODEL + d] : 0.f;
    sA[n] = c32{ val, 0.f };
  }
  fft4096_stockham(sA, sB, tid, -1.f);
  for (int f = tid; f < NFFT; f += 256) Vf[(size_t)d * NFFT + f] = sA[f];
}

// ---------------------------------------------------------------------------
// Per-channel causal conv: FFT(u) * V_f -> iFFT -> 2*Re[0:2048]
// ---------------------------------------------------------------------------
__global__ __launch_bounds__(256)
void conv_fft_kernel(const float* __restrict__ Ut, const c32* __restrict__ Vf,
                     float* __restrict__ Ot) {
  __shared__ c32 sA[NFFT];
  __shared__ c32 sB[NFFT];
  const int c = blockIdx.x;            // b*1024 + d
  const int d = c & 1023;
  const int tid = threadIdx.x;
  const float* u = Ut + (size_t)c * SEQ_LEN;

  for (int n = tid; n < NFFT; n += 256)
    sA[n] = c32{ n < SEQ_LEN ? u[n] : 0.f, 0.f };
  fft4096_stockham(sA, sB, tid, -1.f);

  for (int f = tid; f < NFFT; f += 256) {
    c32 a = sA[f];
    c32 w = Vf[(size_t)d * NFFT + f];
    sA[f] = c32{ a.x * w.x - a.y * w.y, a.x * w.y + a.y * w.x };
  }
  fft4096_stockham(sA, sB, tid, +1.f);

  const float scale = 2.0f / (float)NFFT;   // branch-merge factor 2, iFFT 1/N
  float* o = Ot + (size_t)c * SEQ_LEN;
  for (int t = tid; t < SEQ_LEN; t += 256) o[t] = sA[t].x * scale;
}

// ---------------------------------------------------------------------------
// [b,d,t] -> [b,t,d]
// ---------------------------------------------------------------------------
__global__ void transpose_out_kernel(const float* __restrict__ Ot,
                                     float* __restrict__ out) {
  int idx = blockIdx.x * blockDim.x + threadIdx.x;  // output-linear (coalesced writes)
  int d = idx & 1023;
  int m = idx >> 10;              // b*2048 + t
  int b = m >> 11;
  int t = m & (SEQ_LEN - 1);
  out[idx] = Ot[(((size_t)b << 10) | d) * SEQ_LEN + t];
}

// ---------------------------------------------------------------------------
extern "C" void kernel_launch(void* const* d_in, const int* in_sizes, int n_in,
                              void* d_out, int out_size, void* d_ws, size_t ws_size,
                              hipStream_t stream) {
  const float* x   = (const float*)d_in[0];   // [4,2048,1024]
  const float* Mi  = (const float*)d_in[1];   // [1024,1024]
  const float* Mf  = (const float*)d_in[2];   // [24,1024]
  const float* phi = (const float*)d_in[3];   // [2048,24]
  float* out = (float*)d_out;

  char* w = (char*)d_ws;
  size_t off = 0;
  unsigned short* xb  = (unsigned short*)(w + off); off += (size_t)MROWS * D_MODEL * 2;
  unsigned short* Mbt = (unsigned short*)(w + off); off += (size_t)D_MODEL * D_MODEL * 2;
  float* xproj_t      = (float*)(w + off);          off += (size_t)MROWS * D_MODEL * 4;
  float* V            = (float*)(w + off);          off += (size_t)SEQ_LEN * D_MODEL * 4;
  c32*   Vf           = (c32*)(w + off);            off += (size_t)D_MODEL * NFFT * 8;
  float* Ot           = (float*)(w + off);          off += (size_t)MROWS * D_MODEL * 4;

  cvt_bf16_kernel<<<(MROWS * D_MODEL) / 256, 256, 0, stream>>>(x, xb, MROWS * D_MODEL);
  cvt_transpose_bf16_kernel<<<(D_MODEL * D_MODEL) / 256, 256, 0, stream>>>(Mi, Mbt);

  dim3 gGemm(MROWS / 128, D_MODEL / 128);
  gemm_xproj_wmma<<<gGemm, 128, 0, stream>>>(xb, Mbt, xproj_t);

  phi_proj_kernel<<<(SEQ_LEN * D_MODEL) / 256, 256, 0, stream>>>(phi, Mf, V);
  vf_fft_kernel<<<D_MODEL, 256, 0, stream>>>(V, Vf);
  conv_fft_kernel<<<BATCH * D_MODEL, 256, 0, stream>>>(xproj_t, Vf, Ot);
  transpose_out_kernel<<<(MROWS * D_MODEL) / 256, 256, 0, stream>>>(Ot, out);
}
